// Stickbreaking_76501957476890
// MI455X (gfx1250) — matrix operand order repaired and checked by reading, not compiled, not run on hardware
//
#include <hip/hip_runtime.h>

// Stickbreaking attention, fused flash-style, CDNA5 (gfx1250, wave32, WMMA).
// Shapes: [1, 16, 2048, 64] fp32 in/out.
//
// out[i] = sum_j sigmoid(s_ij) * prod_{k>=j} sigmoid(-s_ik) * v[j]
// Streamed over keys high->low with a per-row log2-space suffix carry.
// Double-buffered LDS staging; DPP row_shl suffix scan; raw exp2/log2.

typedef __attribute__((ext_vector_type(16))) __bf16 v16bf;
typedef __attribute__((ext_vector_type(8)))  __bf16 v8bf;
typedef __attribute__((ext_vector_type(8)))  float  v8f;

#define HEADS 16
#define SEQ   2048
#define HD    64
#define QTILE 128   // query rows per workgroup
#define WAVES 8     // 256 threads, wave32
#define KMB   32    // keys per macro-block (two 16-key WMMA sub-blocks)
#define NB    (SEQ / KMB)

// Padded LDS row strides (in bf16 halves): keep 16B alignment for b128 loads
// while spreading the 16 fragment lanes across the 64 LDS banks.
#define SKP 72      // sK  row stride (key-major, 64 data halves)
#define SVP 40      // sVt row stride (d-major, 32 data halves)
#define SPP 40      // sP  row stride (m-major, 32 data halves)

// Q pre-scale: (1/sqrt(64)) * log2(e)  -> logits come out of WMMA in log2 units
#define QSCALE 0.1803368801111f

// One suffix-scan step: lane i += lane (i+N) within its 16-lane DPP row,
// zero-filled at the row edge (bound_ctrl). row_shl:N == dpp_ctrl 0x100+N.
#define SUFFIX_STEP(sc, CTRL)                                                  \
  sc += __builtin_bit_cast(float, __builtin_amdgcn_update_dpp(                 \
            0, __builtin_bit_cast(int, sc), (CTRL), 0xf, 0xf, true))

// 16 consecutive bf16 halves as a WMMA operand via two 16B LDS loads.
__device__ __forceinline__ v16bf ld16(const __bf16* p) {
  v8bf a = *(const v8bf*)p;
  v8bf b = *(const v8bf*)(p + 8);
  v16bf r;
  #pragma unroll
  for (int i = 0; i < 8; ++i) { r[i] = a[i]; r[8 + i] = b[i]; }
  return r;
}

__global__ __launch_bounds__(256)
void stickbreak_attn(const float* __restrict__ Q,
                     const float* __restrict__ K,
                     const float* __restrict__ V,
                     float* __restrict__ O) {
  __shared__ __attribute__((aligned(64))) __bf16 sK [2][KMB * SKP];     // [key][d]
  __shared__ __attribute__((aligned(64))) __bf16 sVt[2][HD * SVP];      // [d][key]
  __shared__ __attribute__((aligned(64))) __bf16 sP [WAVES][16 * SPP];  // [m][key]

  const int tid  = threadIdx.x;
  const int wv   = tid >> 5;
  const int lane = tid & 31;
  const int n    = lane & 15;   // B/C column index; A row index
  const int hi   = lane >> 4;   // half-wave select

  const int h    = blockIdx.x >> 4;   // head
  const int qt   = blockIdx.x & 15;   // query tile
  const int row0 = qt * QTILE + wv * 16;

  const float* __restrict__ gQ = Q + (size_t)h * SEQ * HD;
  const float* __restrict__ gK = K + (size_t)h * SEQ * HD;
  const float* __restrict__ gV = V + (size_t)h * SEQ * HD;
  float*       __restrict__ gO = O + (size_t)h * SEQ * HD;

  // ---- Q A-fragments: 16x64 bf16 as two K=32 operands, pre-scaled -----
  // A layout (16-bit, 16x32): lane m (=n), halves 0..15 <- k = 16*hi+0..15.
  // Operand o covers d in [32o, 32o+32).
  v16bf qa0, qa1;
  {
    const float* qrow = gQ + (size_t)(row0 + n) * HD;
    #pragma unroll
    for (int o = 0; o < 2; ++o) {
      const float* p0 = qrow + 32 * o + 16 * hi;
      const float* p1 = p0 + 8;
      v16bf a;
      #pragma unroll
      for (int i = 0; i < 8; ++i) {
        a[i]     = (__bf16)(p0[i] * QSCALE);
        a[8 + i] = (__bf16)(p1[i] * QSCALE);
      }
      if (o == 0) qa0 = a; else qa1 = a;
    }
  }

  // ---- prologue: stage last key block into buffer 0 -------------------
  float kreg[8], vreg[8];
  #pragma unroll
  for (int i = 0; i < 8; ++i) {
    int idx = tid + i * 256;                 // idx = key*64 + d, contiguous
    kreg[i] = gK[(size_t)(NB - 1) * KMB * HD + idx];
    vreg[i] = gV[(size_t)(NB - 1) * KMB * HD + idx];
  }
  #pragma unroll
  for (int i = 0; i < 8; ++i) {
    int idx = tid + i * 256;
    int key = idx >> 6, d = idx & 63;
    sK [0][key * SKP + d]   = (__bf16)kreg[i];
    sVt[0][d * SVP + key]   = (__bf16)vreg[i];
  }
  __syncthreads();

  v8f acc0 = {}, acc1 = {}, acc2 = {}, acc3 = {};
  float carry[8];               // running suffix sum of log2(beta) per row
  #pragma unroll
  for (int r = 0; r < 8; ++r) carry[r] = 0.0f;

  int cur = 0;
  for (int b = NB - 1; b >= 0; --b) {
    // ---- issue next block's global loads early (latency under compute)
    if (b > 0) {
      const size_t nbase = (size_t)(b - 1) * KMB * HD;
      #pragma unroll
      for (int i = 0; i < 8; ++i) {
        int idx = tid + i * 256;
        kreg[i] = gK[nbase + idx];
        vreg[i] = gV[nbase + idx];
      }
      if (b > 1) {  // warm L2 two blocks ahead -> global_prefetch_b8
        __builtin_prefetch(gK + (size_t)(b - 2) * KMB * HD + tid * 8, 0, 0);
        __builtin_prefetch(gV + (size_t)(b - 2) * KMB * HD + tid * 8, 0, 0);
      }
    }

    // ---- S = (Q*qscale) * K^T for both 16-key sub-blocks --------------
    // B layout (16-bit, 32x16): lane n = column (key), halves <- k = 16*hi+0..15.
    const __bf16* kb = &sK[cur][0];
    v8f S0 = {}, S1 = {};
    S0 = __builtin_amdgcn_wmma_f32_16x16x32_bf16(false, qa0, false,
           ld16(kb + (0 * 16 + n) * SKP +  0 + 16 * hi), (short)0, S0, false, false);
    S0 = __builtin_amdgcn_wmma_f32_16x16x32_bf16(false, qa1, false,
           ld16(kb + (0 * 16 + n) * SKP + 32 + 16 * hi), (short)0, S0, false, false);
    S1 = __builtin_amdgcn_wmma_f32_16x16x32_bf16(false, qa0, false,
           ld16(kb + (1 * 16 + n) * SKP +  0 + 16 * hi), (short)0, S1, false, false);
    S1 = __builtin_amdgcn_wmma_f32_16x16x32_bf16(false, qa1, false,
           ld16(kb + (1 * 16 + n) * SKP + 32 + 16 * hi), (short)0, S1, false, false);

    // ---- pointwise + per-row inclusive suffix scan (high keys first) --
    // C layout: lane(n,hi), VGPR r <-> row m = r+8*hi, column n (key).
    // log2-space math; raw v_exp_f32/v_log_f32 are exp2/log2 and their
    // arguments here never hit the denormal-fixup range.
    __bf16* pw = &sP[wv][0];
    #pragma unroll
    for (int s = 1; s >= 0; --s) {
      #pragma unroll
      for (int r = 0; r < 8; ++r) {
        float x  = s ? S1[r] : S0[r];                 // log2-scaled logit
        // L = log2(1 + 2^(-|x|))  (arg in [1,2] -> raw log2 is safe)
        float L  = __builtin_amdgcn_logf(1.0f + __builtin_amdgcn_exp2f(-__builtin_fabsf(x)));
        float lb = -(fmaxf(x, 0.0f) + L);             // log2(sigmoid(-x))
        float sc = lb;                                // 16-key suffix scan (DPP)
        SUFFIX_STEP(sc, 0x101);                       // row_shl:1
        SUFFIX_STEP(sc, 0x102);                       // row_shl:2
        SUFFIX_STEP(sc, 0x104);                       // row_shl:4
        SUFFIX_STEP(sc, 0x108);                       // row_shl:8
        float tot = __shfl(sc, 0, 16);                // sub-block total (lane n==0)
        // weight = 2^( log2(sigmoid(x)) + sc + carry ); log2sig(x) = x + lb.
        float wgt = __builtin_amdgcn_exp2f(x + lb + sc + carry[r]);
        carry[r] += tot;
        pw[(r + 8 * hi) * SPP + s * 16 + n] = (__bf16)wgt;
      }
    }
    asm volatile("s_wait_dscnt 0" ::: "memory"); // same-wave LDS RAW

    // ---- P A-fragment from LDS (C-layout -> A-layout re-stripe) -------
    v16bf pa;
    {
      v8bf a0 = *(const v8bf*)&pw[n * SPP + 16 * hi];
      v8bf a1 = *(const v8bf*)&pw[n * SPP + 16 * hi + 8];
      #pragma unroll
      for (int i = 0; i < 8; ++i) { pa[i] = a0[i]; pa[8 + i] = a1[i]; }
    }

    // ---- O += P (16x32) * V (32x64), four 16-wide d chunks ------------
    const __bf16* vb = &sVt[cur][0];
    acc0 = __builtin_amdgcn_wmma_f32_16x16x32_bf16(false, pa, false,
             ld16(vb + (0 * 16 + n) * SVP + 16 * hi), (short)0, acc0, false, false);
    acc1 = __builtin_amdgcn_wmma_f32_16x16x32_bf16(false, pa, false,
             ld16(vb + (1 * 16 + n) * SVP + 16 * hi), (short)0, acc1, false, false);
    acc2 = __builtin_amdgcn_wmma_f32_16x16x32_bf16(false, pa, false,
             ld16(vb + (2 * 16 + n) * SVP + 16 * hi), (short)0, acc2, false, false);
    acc3 = __builtin_amdgcn_wmma_f32_16x16x32_bf16(false, pa, false,
             ld16(vb + (3 * 16 + n) * SVP + 16 * hi), (short)0, acc3, false, false);

    // ---- convert+store next block into the other buffer, then sync ----
    if (b > 0) {
      const int nxt = cur ^ 1;
      #pragma unroll
      for (int i = 0; i < 8; ++i) {
        int idx = tid + i * 256;
        int key = idx >> 6, d = idx & 63;
        sK [nxt][key * SKP + d] = (__bf16)kreg[i];
        sVt[nxt][d * SVP + key] = (__bf16)vreg[i];
      }
      __syncthreads();
      cur = nxt;
    }
  }

  // ---- write output: lane holds rows r+8*hi, columns 16c+n ------------
  #pragma unroll
  for (int r = 0; r < 8; ++r) {
    const size_t row = (size_t)(row0 + r + 8 * hi) * HD;
    gO[row +  0 + n] = acc0[r];
    gO[row + 16 + n] = acc1[r];
    gO[row + 32 + n] = acc2[r];
    gO[row + 48 + n] = acc3[r];
  }
}

extern "C" void kernel_launch(void* const* d_in, const int* in_sizes, int n_in,
                              void* d_out, int out_size, void* d_ws, size_t ws_size,
                              hipStream_t stream) {
  const float* q = (const float*)d_in[0];
  const float* k = (const float*)d_in[1];
  const float* v = (const float*)d_in[2];
  float*       o = (float*)d_out;
  dim3 grid(HEADS * (SEQ / QTILE));  // 256 workgroups
  dim3 block(256);                   // 8 waves (wave32)
  hipLaunchKernelGGL(stickbreak_attn, grid, block, 0, stream, q, k, v, o);
}
